// AnomalyAttention_4810363372538
// MI455X (gfx1250) — matrix operand order repaired and checked by reading, not compiled, not run on hardware
//
#include <hip/hip_runtime.h>
#include <math.h>

typedef __attribute__((ext_vector_type(16))) _Float16 v16h;
typedef __attribute__((ext_vector_type(8)))  float    v8f;
typedef __attribute__((ext_vector_type(4)))  float    v4f;

#define Bn 16
#define Ln 512
#define Hn 8
#define En 64
#define Dn 64
#define SCALE 0.125f          // 1/sqrt(E)
#define LDSW 520              // padded row stride (floats); 520*4 is 16B-aligned

// ---------------------------------------------------------------------------
// Attention: one wave32 per (b, h, 16-row tile).
//   S[16,512] = Q_tile[16,64] @ K[512,64]^T   (v_wmma_f32_16x16x32_f16 x64)
//   softmax rows (normalization deferred) -> series (NT stores)
//   O[16,64]  = inv[m] * (E[16,512] @ V[512,64])  (v_wmma_f32_16x16x32_f16 x64)
// ---------------------------------------------------------------------------
__global__ __launch_bounds__(32) void anomaly_attn_kernel(
    const float* __restrict__ Q, const float* __restrict__ K,
    const float* __restrict__ V, float* __restrict__ Vout,
    float* __restrict__ series)
{
  __shared__ float sP[16][LDSW];   // 16 rows x 512 scores -> exp values (f32)
  __shared__ float sRedMax[2][16];
  __shared__ float sRedSum[2][16];
  __shared__ float sInv[16];       // 1/rowsum

  const int bid  = blockIdx.x;
  const int tile = bid & 31;
  const int h    = (bid >> 5) & 7;
  const int b    = bid >> 8;
  const int row0 = tile * 16;

  const int lane  = threadIdx.x;        // wave32
  const int mrow  = lane & 15;          // A: M index held by this lane
  const int hi    = lane >> 4;          // lane half
  const int kbA   = hi * 16;            // A fragment K base (contiguous, matches B)
  const int koB   = hi * 16;            // B fragment K base
  const int ncol  = lane & 15;          // B/C: N index held by this lane

  // ---- Q tile -> two f16 A-fragments (16x32 each, K=0..31 / 32..63) ----
  const float* qrow = Q + (((size_t)b * Ln + row0 + mrow) * Hn + h) * En;
  v16h a0, a1;
#pragma unroll
  for (int i = 0; i < 8; ++i) {
    a0[i]     = (_Float16)qrow[kbA + i];
    a0[8 + i] = (_Float16)qrow[kbA + 8 + i];
    a1[i]     = (_Float16)qrow[32 + kbA + i];
    a1[8 + i] = (_Float16)qrow[32 + kbA + 8 + i];
  }

  // ---- scores: 32 column tiles of 16 ----
  for (int t = 0; t < 32; ++t) {
    const int s0 = t * 16;
    const float* krow = K + (((size_t)b * Ln + s0 + ncol) * Hn + h) * En;
    v16h b0, b1;
#pragma unroll
    for (int i = 0; i < 16; ++i) {
      b0[i] = (_Float16)krow[koB + i];        // e = 0..31 across lane halves
      b1[i] = (_Float16)krow[32 + koB + i];   // e = 32..63
    }
    v8f c = {};
    c = __builtin_amdgcn_wmma_f32_16x16x32_f16(false, a0, false, b0,
                                               (short)0, c, false, false);
    c = __builtin_amdgcn_wmma_f32_16x16x32_f16(false, a1, false, b1,
                                               (short)0, c, false, false);
#pragma unroll
    for (int r = 0; r < 8; ++r) {             // C: VGPR r -> M = r + 8*hi
      const int mm = r + hi * 8;
      const int ss = s0 + ncol;
      float v = c[r] * SCALE;
      if (ss > row0 + mm) v = -__builtin_inff();   // causal mask
      sP[mm][ss] = v;
    }
  }
  __syncthreads();

  // ---- softmax (vectorized, normalization deferred): 2 lanes per row ----
  {
    const int row  = lane & 15;
    const int half = lane >> 4;
    v4f* rp = (v4f*)&sP[row][half * 256];
    float mx = -__builtin_inff();
    for (int j = 0; j < 64; ++j) {
      v4f x = rp[j];
      mx = fmaxf(mx, fmaxf(fmaxf(x.x, x.y), fmaxf(x.z, x.w)));
    }
    sRedMax[half][row] = mx;
    __syncthreads();
    const float rmx = fmaxf(sRedMax[0][row], sRedMax[1][row]);
    float sum = 0.f;
    for (int j = 0; j < 64; ++j) {
      v4f x = rp[j];
      x.x = __expf(x.x - rmx);
      x.y = __expf(x.y - rmx);
      x.z = __expf(x.z - rmx);
      x.w = __expf(x.w - rmx);
      rp[j] = x;
      sum += x.x + x.y + x.z + x.w;
    }
    sRedSum[half][row] = sum;
    __syncthreads();
    sInv[row] = 1.0f / (sRedSum[0][row] + sRedSum[1][row]);  // both halves agree
    __syncthreads();
  }

  // ---- write series tile: 128B NT lines, scale on the fly ----
  {
    float* srow = series + (((size_t)b * Hn + h) * Ln + row0) * (size_t)Ln;
#pragma unroll 2
    for (int r = 0; r < 16; ++r) {
      const float invr = sInv[r];
#pragma unroll
      for (int cch = 0; cch < 4; ++cch) {
        const int s = cch * 128 + lane * 4;
        v4f x = *(v4f*)&sP[r][s];
        x *= invr;
        __builtin_nontemporal_store(x, (v4f*)&srow[(size_t)r * Ln + s]);
      }
    }
  }

  // ---- O = E @ V : 16 K-chunks x 4 D-tiles (normalize at epilogue) ----
  v8f acc[4] = { {}, {}, {}, {} };
  for (int kc = 0; kc < 16; ++kc) {
    const int kb = kc * 32;
    v16h pa;
#pragma unroll
    for (int i = 0; i < 8; ++i) {
      pa[i]     = (_Float16)sP[mrow][kb + kbA + i];
      pa[8 + i] = (_Float16)sP[mrow][kb + kbA + 8 + i];
    }
    const float* vcol = V + (((size_t)b * Ln + kb + koB) * Hn + h) * Dn + ncol;
#pragma unroll
    for (int d = 0; d < 4; ++d) {
      v16h vb;
#pragma unroll
      for (int i = 0; i < 16; ++i)
        vb[i] = (_Float16)vcol[(size_t)i * (Hn * Dn) + d * 16];
      acc[d] = __builtin_amdgcn_wmma_f32_16x16x32_f16(false, pa, false, vb,
                                                      (short)0, acc[d],
                                                      false, false);
    }
  }
#pragma unroll
  for (int r = 0; r < 8; ++r) {
    const int mm = r + hi * 8;
    const float invm = sInv[mm];
    const size_t obase = (((size_t)b * Ln + row0 + mm) * Hn + h) * Dn + ncol;
#pragma unroll
    for (int d = 0; d < 4; ++d)
      __builtin_nontemporal_store(acc[d][r] * invm, &Vout[obase + d * 16]);
  }
}

// ---------------------------------------------------------------------------
// Gaussian prior + sigma broadcast: one 128-thread block per (b,h,l) row.
// float4 NT streaming writes; one v_exp_f32 per element.
// ---------------------------------------------------------------------------
__global__ __launch_bounds__(128) void anomaly_prior_kernel(
    const float* __restrict__ sigma, float* __restrict__ prior,
    float* __restrict__ sigfull)
{
  const int bid = blockIdx.x;
  const int l = bid & (Ln - 1);
  const int h = (bid >> 9) & 7;
  const int b = bid >> 12;

  const float x  = sigma[((size_t)b * Ln + l) * Hn + h];
  float sg = 1.0f / (1.0f + __expf(-5.0f * x)) + 1e-5f;  // sigmoid(5x)+1e-5
  sg = __expf(1.0986122886681098f * sg) - 1.0f;          // 3^sg - 1
  const float amp = 0.3989422804014327f / sg;            // 1/sqrt(2pi)/sg
  const float c2  = -0.5f / (sg * sg);

  const size_t rowoff = (((size_t)b * Hn + h) * Ln + l) * (size_t)Ln;
  const int s0 = (int)threadIdx.x * 4;

  v4f p, sgv;
#pragma unroll
  for (int j = 0; j < 4; ++j) {
    const float d = (float)(l - (s0 + j));
    p[j]   = amp * __expf(d * d * c2);
    sgv[j] = sg;
  }
  __builtin_nontemporal_store(p,   (v4f*)&prior[rowoff + s0]);
  __builtin_nontemporal_store(sgv, (v4f*)&sigfull[rowoff + s0]);
}

// ---------------------------------------------------------------------------
extern "C" void kernel_launch(void* const* d_in, const int* in_sizes, int n_in,
                              void* d_out, int out_size, void* d_ws,
                              size_t ws_size, hipStream_t stream) {
  (void)in_sizes; (void)n_in; (void)out_size; (void)d_ws; (void)ws_size;
  const float* Q   = (const float*)d_in[0];
  const float* K   = (const float*)d_in[1];
  const float* V   = (const float*)d_in[2];
  const float* sig = (const float*)d_in[3];
  // d_in[4] (attn_mask) is the known causal triangle -> derived from indices.

  float* out     = (float*)d_out;
  float* Vout    = out;                                      // [B,L,H,D]
  float* series  = out + (size_t)Bn * Ln * Hn * Dn;          // [B,H,L,L]
  float* prior   = series + (size_t)Bn * Hn * Ln * Ln;       // [B,H,L,L]
  float* sigfull = prior + (size_t)Bn * Hn * Ln * Ln;        // [B,H,L,L]

  anomaly_attn_kernel<<<Bn * Hn * (Ln / 16), 32, 0, stream>>>(Q, K, V, Vout,
                                                              series);
  anomaly_prior_kernel<<<Bn * Hn * Ln, 128, 0, stream>>>(sig, prior, sigfull);
}
